// SchnetConv_25220047962133
// MI455X (gfx1250) — matrix operand order repaired and checked by reading, not compiled, not run on hardware
//
#include <hip/hip_runtime.h>

// ---------------------------------------------------------------------------
// SchNet cfconv fused kernels for MI455X (gfx1250, wave32, WMMA bf16)
//   edge kernel:  w = ssp(bf@W1+b1); w = ssp(w@W2+b2); m = node_h[src]*w*edge_h
//                 atomic scatter-add into sAcc[dst], cnt[dst]
//   node kernel:  h = sAcc/max(cnt,1); out = ssp(ssp(h@W3+b3)@W4+b4)
// Matrix math in V_WMMA_F32_16X16X32_BF16 (f32 accumulate).
// ---------------------------------------------------------------------------

typedef __attribute__((ext_vector_type(16))) __bf16 v16bf;
typedef __attribute__((ext_vector_type(8)))  __bf16 v8bf;
typedef __attribute__((ext_vector_type(8)))  float  v8f;
typedef __attribute__((ext_vector_type(4)))  float  f32x4;

#define F_IN 64
#define RAD  128

// Native f32 -> bf16 conversion: let the backend select hardware converts
// (v_cvt_pk_bf16_f32 packs two per op, matching the fragment packing).
__device__ __forceinline__ __bf16 f2bf(float f) { return (__bf16)f; }
__device__ __forceinline__ unsigned short f2bfbits(float f) {
  return __builtin_bit_cast(unsigned short, (__bf16)f);
}

__device__ __forceinline__ float ssp(float x) {
  // shifted softplus: log(1+exp(x)) - log(2), numerically stable
  float ax = __builtin_fabsf(x);
  return fmaxf(x, 0.0f) + __logf(1.0f + __expf(-ax)) - 0.69314718055994531f;
}

// 16 contiguous bf16 elements -> one fragment
__device__ __forceinline__ v16bf ld_frag16(const unsigned short* p) {
  v8bf lo = *(const v8bf*)(p);
  v8bf hi = *(const v8bf*)(p + 8);
  return __builtin_shufflevector(lo, hi, 0,1,2,3,4,5,6,7,8,9,10,11,12,13,14,15);
}
// two disjoint 8-element bf16 runs -> one fragment (A-matrix layout)
__device__ __forceinline__ v16bf ld_frag_split(const unsigned short* p0,
                                               const unsigned short* p1) {
  v8bf lo = *(const v8bf*)(p0);
  v8bf hi = *(const v8bf*)(p1);
  return __builtin_shufflevector(lo, hi, 0,1,2,3,4,5,6,7,8,9,10,11,12,13,14,15);
}

// ---------------------------------------------------------------------------
__global__ void __launch_bounds__(256)
schnet_edge_kernel(const float* __restrict__ node_h,
                   const float* __restrict__ edge_h,
                   const float* __restrict__ bfeat,
                   const float* __restrict__ W1, const float* __restrict__ b1,
                   const float* __restrict__ W2, const float* __restrict__ b2,
                   const int*   __restrict__ src, const int* __restrict__ dst,
                   float* __restrict__ sAcc, float* __restrict__ cnt,
                   int nEdges)
{
  __shared__ __align__(16) unsigned short W1T[F_IN * RAD];    // [n][k] bf16, 16 KB
  __shared__ __align__(16) unsigned short W2T[F_IN * F_IN];   // [n][k] bf16,  8 KB
  __shared__ __align__(16) unsigned short scr[8][16 * F_IN];  // wave-private 16x64

  // Stage weights transposed into LDS so each B-fragment is two contiguous 8-element k runs.
  for (int i = threadIdx.x; i < RAD * F_IN; i += 256) {
    int k = i >> 6, n = i & 63;
    W1T[n * RAD + k] = f2bfbits(W1[i]);
  }
  for (int i = threadIdx.x; i < F_IN * F_IN; i += 256) {
    int k = i >> 6, n = i & 63;
    W2T[n * F_IN + k] = f2bfbits(W2[i]);
  }
  __syncthreads();

  const int lane = threadIdx.x & 31;
  const int hl   = lane & 15;            // half-lane (M or N index)
  const int hi   = lane >> 4;            // lane group 0/1
  const int slot = threadIdx.x >> 5;     // wave index in block
  const int wid    = (blockIdx.x * blockDim.x + threadIdx.x) >> 5;
  const int nWaves = (gridDim.x * blockDim.x) >> 5;
  const int nTiles = (nEdges + 15) >> 4;

  float b1v[4], b2v[4];
  #pragma unroll
  for (int nt = 0; nt < 4; ++nt) {
    b1v[nt] = b1[nt * 16 + hl];
    b2v[nt] = b2[nt * 16 + hl];
  }

  for (int tile = wid; tile < nTiles; tile += nWaves) {
    const int base = tile << 4;
    const int rowE = min(base + hl, nEdges - 1);

    // Prefetch next tile's radial features (global_prefetch_b8).
    {
      int nb = base + (nWaves << 4);
      if (nb < nEdges)
        __builtin_prefetch(&bfeat[(size_t)(nb + hl) * RAD + hi * 64], 0, 1);
    }

    // ---- GEMM1: acc = bf @ W1, K = 128 (4 chunks of 32) -------------------
    v8f acc[4] = {};
    #pragma unroll
    for (int kk = 0; kk < 4; ++kk) {
      // A fragment: e<8 -> K = kk*32 + hi*8 + e ; e>=8 -> +16 more
      const float* pA = &bfeat[(size_t)rowE * RAD + kk * 32 + hi * 8];
      f32x4 x0 = __builtin_nontemporal_load((const f32x4*)(pA));
      f32x4 x1 = __builtin_nontemporal_load((const f32x4*)(pA + 4));
      f32x4 y0 = __builtin_nontemporal_load((const f32x4*)(pA + 16));
      f32x4 y1 = __builtin_nontemporal_load((const f32x4*)(pA + 20));
      v16bf a;
      #pragma unroll
      for (int j = 0; j < 4; ++j) {
        a[j]      = f2bf(x0[j]);
        a[4 + j]  = f2bf(x1[j]);
        a[8 + j]  = f2bf(y0[j]);
        a[12 + j] = f2bf(y1[j]);
      }
      #pragma unroll
      for (int nt = 0; nt < 4; ++nt) {
        v16bf b = ld_frag_split(&W1T[(nt * 16 + hl) * RAD + kk * 32 + hi * 8],
                                &W1T[(nt * 16 + hl) * RAD + kk * 32 + 16 + hi * 8]);
        acc[nt] = __builtin_amdgcn_wmma_f32_16x16x32_bf16(
            false, a, false, b, (short)0, acc[nt], false, false);
      }
    }

    // bias + ssp, then C/D layout -> A layout via wave-private LDS scratch
    #pragma unroll
    for (int nt = 0; nt < 4; ++nt) {
      #pragma unroll
      for (int r = 0; r < 8; ++r) {
        int row = hi ? (r + 8) : r;   // C/D: lanes16-31 hold M = r+8
        scr[slot][row * F_IN + nt * 16 + hl] = f2bfbits(ssp(acc[nt][r] + b1v[nt]));
      }
    }
    // scratch is wave-private; same-wave LDS ops are in order -> no barrier

    // ---- GEMM2: acc2 = w @ W2, K = 64 (2 chunks of 32) --------------------
    v8f acc2[4] = {};
    #pragma unroll
    for (int kk = 0; kk < 2; ++kk) {
      v16bf a = ld_frag_split(&scr[slot][hl * F_IN + kk * 32 + hi * 8],
                              &scr[slot][hl * F_IN + kk * 32 + 16 + hi * 8]);
      #pragma unroll
      for (int nt = 0; nt < 4; ++nt) {
        v16bf b = ld_frag_split(&W2T[(nt * 16 + hl) * F_IN + kk * 32 + hi * 8],
                                &W2T[(nt * 16 + hl) * F_IN + kk * 32 + 16 + hi * 8]);
        acc2[nt] = __builtin_amdgcn_wmma_f32_16x16x32_bf16(
            false, a, false, b, (short)0, acc2[nt], false, false);
      }
    }

    // ---- gate, gather, scatter-add (segment sum lives in L2) --------------
    #pragma unroll
    for (int r = 0; r < 8; ++r) {
      int e = base + (hi ? r + 8 : r);
      if (e < nEdges) {
        int si = src[e], di = dst[e];
        #pragma unroll
        for (int nt = 0; nt < 4; ++nt) {
          int feat = nt * 16 + hl;
          float w2v = ssp(acc2[nt][r] + b2v[nt]);
          float ehv = w2v * __builtin_nontemporal_load(&edge_h[(size_t)e * F_IN + feat]);
          float mv  = node_h[(size_t)si * F_IN + feat] * ehv;
          atomicAdd(&sAcc[(size_t)di * F_IN + feat], mv);
        }
      }
    }
    if (lane < 16) {
      int e = base + lane;
      if (e < nEdges) atomicAdd(&cnt[dst[e]], 1.0f);
    }
  }
}

// ---------------------------------------------------------------------------
__global__ void __launch_bounds__(256)
schnet_node_kernel(const float* __restrict__ sAcc, const float* __restrict__ cnt,
                   const float* __restrict__ W3, const float* __restrict__ b3,
                   const float* __restrict__ W4, const float* __restrict__ b4,
                   float* __restrict__ out, int nNodes)
{
  __shared__ __align__(16) unsigned short W3T[F_IN * F_IN];
  __shared__ __align__(16) unsigned short W4T[F_IN * F_IN];
  __shared__ __align__(16) unsigned short scr[8][16 * F_IN];

  for (int i = threadIdx.x; i < F_IN * F_IN; i += 256) {
    int k = i >> 6, n = i & 63;
    W3T[n * F_IN + k] = f2bfbits(W3[i]);
    W4T[n * F_IN + k] = f2bfbits(W4[i]);
  }
  __syncthreads();

  const int lane = threadIdx.x & 31;
  const int hl   = lane & 15;
  const int hi   = lane >> 4;
  const int slot = threadIdx.x >> 5;
  const int wid    = (blockIdx.x * blockDim.x + threadIdx.x) >> 5;
  const int nWaves = (gridDim.x * blockDim.x) >> 5;
  const int nTiles = (nNodes + 15) >> 4;

  float b3v[4], b4v[4];
  #pragma unroll
  for (int nt = 0; nt < 4; ++nt) {
    b3v[nt] = b3[nt * 16 + hl];
    b4v[nt] = b4[nt * 16 + hl];
  }

  for (int tile = wid; tile < nTiles; tile += nWaves) {
    const int base = tile << 4;
    const int row  = min(base + hl, nNodes - 1);
    const float inv = 1.0f / fmaxf(cnt[row], 1.0f);

    // ---- GEMM3: acc = (sAcc/cnt) @ W3, K = 64 -----------------------------
    v8f acc[4] = {};
    #pragma unroll
    for (int kk = 0; kk < 2; ++kk) {
      const float* pA = &sAcc[(size_t)row * F_IN + kk * 32 + hi * 8];
      f32x4 x0 = *(const f32x4*)(pA);
      f32x4 x1 = *(const f32x4*)(pA + 4);
      f32x4 y0 = *(const f32x4*)(pA + 16);
      f32x4 y1 = *(const f32x4*)(pA + 20);
      v16bf a;
      #pragma unroll
      for (int j = 0; j < 4; ++j) {
        a[j]      = f2bf(x0[j] * inv);
        a[4 + j]  = f2bf(x1[j] * inv);
        a[8 + j]  = f2bf(y0[j] * inv);
        a[12 + j] = f2bf(y1[j] * inv);
      }
      #pragma unroll
      for (int nt = 0; nt < 4; ++nt) {
        v16bf b = ld_frag_split(&W3T[(nt * 16 + hl) * F_IN + kk * 32 + hi * 8],
                                &W3T[(nt * 16 + hl) * F_IN + kk * 32 + 16 + hi * 8]);
        acc[nt] = __builtin_amdgcn_wmma_f32_16x16x32_bf16(
            false, a, false, b, (short)0, acc[nt], false, false);
      }
    }

    #pragma unroll
    for (int nt = 0; nt < 4; ++nt) {
      #pragma unroll
      for (int r = 0; r < 8; ++r) {
        int rr = hi ? (r + 8) : r;
        scr[slot][rr * F_IN + nt * 16 + hl] = f2bfbits(ssp(acc[nt][r] + b3v[nt]));
      }
    }

    // ---- GEMM4: acc2 = ssp(...) @ W4, K = 64 ------------------------------
    v8f acc2[4] = {};
    #pragma unroll
    for (int kk = 0; kk < 2; ++kk) {
      v16bf a = ld_frag_split(&scr[slot][hl * F_IN + kk * 32 + hi * 8],
                              &scr[slot][hl * F_IN + kk * 32 + 16 + hi * 8]);
      #pragma unroll
      for (int nt = 0; nt < 4; ++nt) {
        v16bf b = ld_frag_split(&W4T[(nt * 16 + hl) * F_IN + kk * 32 + hi * 8],
                                &W4T[(nt * 16 + hl) * F_IN + kk * 32 + 16 + hi * 8]);
        acc2[nt] = __builtin_amdgcn_wmma_f32_16x16x32_bf16(
            false, a, false, b, (short)0, acc2[nt], false, false);
      }
    }

    #pragma unroll
    for (int r = 0; r < 8; ++r) {
      int n = base + (hi ? r + 8 : r);
      if (n < nNodes) {
        #pragma unroll
        for (int nt = 0; nt < 4; ++nt) {
          out[(size_t)n * F_IN + nt * 16 + hl] = ssp(acc2[nt][r] + b4v[nt]);
        }
      }
    }
  }
}

// ---------------------------------------------------------------------------
__global__ void zero_kernel(float* __restrict__ p, long long n) {
  long long i = (long long)blockIdx.x * blockDim.x + threadIdx.x;
  long long stride = (long long)gridDim.x * blockDim.x;
  for (; i < n; i += stride) p[i] = 0.0f;
}

// ---------------------------------------------------------------------------
extern "C" void kernel_launch(void* const* d_in, const int* in_sizes, int n_in,
                              void* d_out, int out_size, void* d_ws, size_t ws_size,
                              hipStream_t stream) {
  (void)n_in; (void)out_size; (void)ws_size;
  const float* node_h = (const float*)d_in[0];
  const float* edge_h = (const float*)d_in[1];
  const float* bfeat  = (const float*)d_in[2];
  const float* W1 = (const float*)d_in[3];
  const float* b1 = (const float*)d_in[4];
  const float* W2 = (const float*)d_in[5];
  const float* b2 = (const float*)d_in[6];
  const float* W3 = (const float*)d_in[7];
  const float* b3 = (const float*)d_in[8];
  const float* W4 = (const float*)d_in[9];
  const float* b4 = (const float*)d_in[10];
  const int* src = (const int*)d_in[11];
  const int* dst = (const int*)d_in[12];

  const int nNodes = in_sizes[0] / F_IN;
  const int nEdges = in_sizes[11];

  float* sAcc = (float*)d_ws;                       // [nNodes, 64] f32
  float* cnt  = sAcc + (size_t)nNodes * F_IN;       // [nNodes]     f32

  long long zn = (long long)nNodes * F_IN + nNodes;
  zero_kernel<<<512, 256, 0, stream>>>(sAcc, zn);

  schnet_edge_kernel<<<1024, 256, 0, stream>>>(node_h, edge_h, bfeat,
                                               W1, b1, W2, b2, src, dst,
                                               sAcc, cnt, nEdges);

  int nodeTiles  = (nNodes + 15) / 16;
  int nodeBlocks = (nodeTiles + 7) / 8;             // 8 waves per block
  schnet_node_kernel<<<nodeBlocks, 256, 0, stream>>>(sAcc, cnt, W3, b3, W4, b4,
                                                     (float*)d_out, nNodes);
}
